// DisenCDR_50560355009134
// MI455X (gfx1250) — matrix operand order repaired and checked by reading, not compiled, not run on hardware
//
#include <hip/hip_runtime.h>
#include <hip/hip_bf16.h>

// ---------------------------------------------------------------------------
// DisenCDR cross-domain GCN forward for MI455X (gfx1250, wave32, WMMA).
//   - GEMMs via v_wmma_f32_16x16x32_f16 (f16 operands, f32 accumulate)
//   - SpMM via one-wave-per-edge float4 gather + native f32 global atomics
//   - elementwise bias/act/fuse passes in float4
// ---------------------------------------------------------------------------

typedef __attribute__((ext_vector_type(16))) _Float16 v16h;
typedef __attribute__((ext_vector_type(8)))  float    v8f;

#define ACT_NONE  0
#define ACT_LEAKY 1   // leaky_relu, alpha = 0.2
#define ACT_RELU  2

#define ALPHA 0.2f
#define RATE  0.7f

#define NI_CONST 200000   // items per domain (fixed by the model)

__device__ __forceinline__ void atomic_add_f32(float* p, float v) {
  // native global_atomic_add_f32 (no return) on AMD
  unsafeAtomicAdd(p, v);
}

__device__ __forceinline__ float apply_act(float v, int act) {
  if (act == ACT_LEAKY) return (v < 0.f) ? ALPHA * v : v;
  if (act == ACT_RELU)  return (v < 0.f) ? 0.f : v;
  return v;
}

// ---------------------------------------------------------------------------
// Pre-swizzle weights [Kdim,128] fp32 -> f16 B-fragments in WMMA lane layout.
// grid = (8 n-tiles, Kdim/32 k-blocks), block = 32 (one wave).
// B fragment layout (16-bit B, 32x16): lane = column (mod 16);
// lanes 0-15 hold K=0..7,16..23 of the k-block, lanes 16-31 hold K=8..15,24..31,
// 2 consecutive K per VGPR (v16h element e <-> K = kgroup*8 + (e&7) + (e>>3)*16).
// ---------------------------------------------------------------------------
__global__ void wswz_prep_kernel(const float* __restrict__ W,
                                 _Float16* __restrict__ wswz) {
  const int ntile = blockIdx.x;        // 0..7
  const int kkb   = blockIdx.y;        // 0..Kdim/32-1
  const int lane  = threadIdx.x;       // 0..31
  const int n     = ntile * 16 + (lane & 15);
  const int kbase = (lane >> 4) * 8;
  v16h out;
#pragma unroll
  for (int e = 0; e < 16; ++e) {
    const int k = kkb * 32 + kbase + (e & 7) + ((e >> 3) << 4);
    out[e] = (_Float16)W[(size_t)k * 128 + n];
  }
  *(v16h*)(wswz + (((size_t)ntile * gridDim.y + kkb) * 32 + lane) * 16) = out;
}

// ---------------------------------------------------------------------------
// Y[N,128] = act( concat(X0,X1)[N,Kdim] @ W[Kdim,128] + bias )
// X1 may be null when Kdim == 128. W comes pre-swizzled (wswz).
// grid.x = N/16 row tiles; block = (32, 8): one wave per (rowtile, ntile).
// ---------------------------------------------------------------------------
__global__ void gemm_wmma_kernel(const float* __restrict__ X0,
                                 const float* __restrict__ X1,
                                 const _Float16* __restrict__ wswz,
                                 const float* __restrict__ bias,
                                 float* __restrict__ Y,
                                 int N, int Kdim, int act) {
  const int lane  = threadIdx.x;           // 0..31
  const int ntile = threadIdx.y;           // 0..7
  const int m0    = blockIdx.x * 16;
  const int kkblocks = Kdim >> 5;

  int m = m0 + (lane & 15);
  if (m >= N) m = N - 1;                   // N is a multiple of 16 here anyway
  const int khalf = (lane >> 4) * 8;       // A frag: lanes16-31 offset K by +8

  v8f c = {};
  for (int kb = 0; kb < kkblocks; ++kb) {
    const int kk = kb * 32;
    const float* Xs = (kk < 128) ? X0 : X1;        // concat without materializing
    const int kx = kk & 127;

    // A fragment (16-bit A 16x32 layout): per lane two float8 chunks of its row
    const float* xr = Xs + (size_t)m * 128 + kx + khalf;
    const float4 a0 = *(const float4*)(xr);        // K = khalf+0..3   -> e0..3
    const float4 a1 = *(const float4*)(xr + 4);    // K = khalf+4..7   -> e4..7
    const float4 a2 = *(const float4*)(xr + 16);   // K = 16+khalf+0..3-> e8..11
    const float4 a3 = *(const float4*)(xr + 20);   // K = 16+khalf+4..7-> e12..15
    v16h a;
    a[0]  = (_Float16)a0.x; a[1]  = (_Float16)a0.y;
    a[2]  = (_Float16)a0.z; a[3]  = (_Float16)a0.w;
    a[4]  = (_Float16)a1.x; a[5]  = (_Float16)a1.y;
    a[6]  = (_Float16)a1.z; a[7]  = (_Float16)a1.w;
    a[8]  = (_Float16)a2.x; a[9]  = (_Float16)a2.y;
    a[10] = (_Float16)a2.z; a[11] = (_Float16)a2.w;
    a[12] = (_Float16)a3.x; a[13] = (_Float16)a3.y;
    a[14] = (_Float16)a3.z; a[15] = (_Float16)a3.w;

    // B fragment: contiguous 32B pre-swizzled load
    const v16h b = *(const v16h*)(wswz +
        (((size_t)ntile * kkblocks + kb) * 32 + lane) * 16);

    c = __builtin_amdgcn_wmma_f32_16x16x32_f16(
        /*neg_a=*/false, a, /*neg_b=*/false, b,
        /*c_mod=*/(short)0, c, /*reuse_a=*/false, /*reuse_b=*/false);
  }

  // C/D layout: VGPR r -> row M = r + 8*(lane>=16), col = lane&15
  const int col   = ntile * 16 + (lane & 15);
  const int rbase = (lane >> 4) * 8;
  const float bv  = bias ? bias[col] : 0.f;
#pragma unroll
  for (int r = 0; r < 8; ++r) {
    const int mm = m0 + rbase + r;
    if (mm < N) {
      Y[(size_t)mm * 128 + col] = apply_act(c[r] + bv, act);
    }
  }
}

// ---------------------------------------------------------------------------
// Zero fill (float4 granularity). n4 = number of float4s.
// ---------------------------------------------------------------------------
__global__ void fill_zero_kernel(float4* __restrict__ p, long long n4) {
  long long i = (long long)blockIdx.x * blockDim.x + threadIdx.x;
  if (i < n4) p[i] = make_float4(0.f, 0.f, 0.f, 0.f);
}

// ---------------------------------------------------------------------------
// SpMM scatter: Y[rows[e]] += vals[e] * X[cols[e]]  (rows of 128 floats).
// One wave per edge; lane handles 4 features (float4 gather + 4 fp32 atomics).
// ---------------------------------------------------------------------------
__global__ void spmm_scatter_kernel(const int* __restrict__ rows,
                                    const int* __restrict__ cols,
                                    const float* __restrict__ vals,
                                    const float* __restrict__ Xin,
                                    float* __restrict__ Yout, int E) {
  const long long gid = (long long)blockIdx.x * blockDim.x + threadIdx.x;
  const int e = (int)(gid >> 5);
  if (e >= E) return;
  const int lane = (int)(gid & 31);
  const int r = rows[e];
  const int c = cols[e];
  const float v = vals[e];
  const float4 x = *(const float4*)(Xin + (size_t)c * 128 + lane * 4);
  float* yp = Yout + (size_t)r * 128 + lane * 4;
  atomic_add_f32(yp + 0, v * x.x);
  atomic_add_f32(yp + 1, v * x.y);
  atomic_add_f32(yp + 2, v * x.z);
  atomic_add_f32(yp + 3, v * x.w);
}

// ---------------------------------------------------------------------------
// Y[N,128] = act(Y + bias[128])  (float4 granularity)
// ---------------------------------------------------------------------------
__global__ void bias_act_kernel(float4* __restrict__ Y,
                                const float4* __restrict__ bias,
                                long long n4, int act) {
  long long i = (long long)blockIdx.x * blockDim.x + threadIdx.x;
  if (i >= n4) return;
  const float4 b = bias[i & 31];
  float4 y = Y[i];
  y.x = apply_act(y.x + b.x, act);
  y.y = apply_act(y.y + b.y, act);
  y.z = apply_act(y.z + b.z, act);
  y.w = apply_act(y.w + b.w, act);
  Y[i] = y;
}

// ---------------------------------------------------------------------------
// O = RATE*A + (1-RATE)*B  (float4 granularity)
// ---------------------------------------------------------------------------
__global__ void fuse_kernel(const float4* __restrict__ A,
                            const float4* __restrict__ B,
                            float4* __restrict__ O, long long n4) {
  long long i = (long long)blockIdx.x * blockDim.x + threadIdx.x;
  if (i >= n4) return;
  const float4 a = A[i];
  const float4 b = B[i];
  float4 o;
  o.x = RATE * a.x + (1.f - RATE) * b.x;
  o.y = RATE * a.y + (1.f - RATE) * b.y;
  o.z = RATE * a.z + (1.f - RATE) * b.z;
  o.w = RATE * a.w + (1.f - RATE) * b.w;
  O[i] = o;
}

// ---------------------------------------------------------------------------
// Host-side helpers
// ---------------------------------------------------------------------------
static void run_gemm(const float* X0, const float* X1, const float* W,
                     const float* bias, float* Y, int N, int Kdim, int act,
                     _Float16* wswz, hipStream_t s) {
  const int kkb = Kdim / 32;
  wswz_prep_kernel<<<dim3(8, kkb), 32, 0, s>>>(W, wswz);
  gemm_wmma_kernel<<<dim3((N + 15) / 16), dim3(32, 8), 0, s>>>(
      X0, X1, wswz, bias, Y, N, Kdim, act);
}

static void run_scatter_gcn(const int* rows, const int* cols, const float* vals,
                            int E, const float* Xin, float* Yout, int Nout,
                            const float* bias, int act, hipStream_t s) {
  const long long n4 = (long long)Nout * 32;
  fill_zero_kernel<<<(unsigned)((n4 + 255) / 256), 256, 0, s>>>(
      (float4*)Yout, n4);
  const long long tot = (long long)E * 32;
  spmm_scatter_kernel<<<(unsigned)((tot + 255) / 256), 256, 0, s>>>(
      rows, cols, vals, Xin, Yout, E);
  bias_act_kernel<<<(unsigned)((n4 + 255) / 256), 256, 0, s>>>(
      (float4*)Yout, (const float4*)bias, n4, act);
}

static void run_fuse(const float* A, const float* B, float* O, long long nflt,
                     hipStream_t s) {
  const long long n4 = nflt / 4;
  fuse_kernel<<<(unsigned)((n4 + 255) / 256), 256, 0, s>>>(
      (const float4*)A, (const float4*)B, (float4*)O, n4);
}

extern "C" void kernel_launch(void* const* d_in, const int* in_sizes, int n_in,
                              void* d_out, int out_size, void* d_ws, size_t ws_size,
                              hipStream_t stream) {
  (void)n_in; (void)out_size; (void)ws_size;

  const float* s_fea = (const float*)d_in[0];
  const float* t_fea = (const float*)d_in[1];
  const int*   se_u  = (const int*)d_in[2];
  const int*   se_i  = (const int*)d_in[3];
  const float* se_v  = (const float*)d_in[4];
  const int*   te_u  = (const int*)d_in[5];
  const int*   te_i  = (const int*)d_in[6];
  const float* te_v  = (const float*)d_in[7];

  const int Es = in_sizes[2];
  const int Et = in_sizes[5];
  const int NU = in_sizes[0] / 128;
  const int NI = NI_CONST;

  // params, flattened in PARAM_SHAPES order starting at d_in[8]
  const float* prm[32];
  for (int i = 0; i < 32; ++i) prm[i] = (const float*)d_in[8 + i];
  const float *l1_gc1_W = prm[0],  *l1_gc1_b = prm[1];
  const float *l1_gc2_W = prm[2],  *l1_gc2_b = prm[3];
  const float *l1_gc3_W = prm[4],  *l1_gc3_b = prm[5];
  const float *l1_gc4_W = prm[6],  *l1_gc4_b = prm[7];
  const float *l1_su_W  = prm[8],  *l1_su_b  = prm[9];
  const float *l1_tu_W  = prm[10], *l1_tu_b  = prm[11];
  const float *l2_gc1_W = prm[12], *l2_gc1_b = prm[13];
  const float *l2_gc2_W = prm[14], *l2_gc2_b = prm[15];
  const float *l2_gc3m_W = prm[16], *l2_gc3m_b = prm[17];
  const float *l2_gc3s_W = prm[18], *l2_gc3s_b = prm[19];
  const float *l2_gc4m_W = prm[20], *l2_gc4m_b = prm[21];
  const float *l2_gc4s_W = prm[22], *l2_gc4s_b = prm[23];
  const float *l2_sum_W = prm[24], *l2_sum_b = prm[25];
  const float *l2_sus_W = prm[26], *l2_sus_b = prm[27];
  const float *l2_tum_W = prm[28], *l2_tum_b = prm[29];
  const float *l2_tus_W = prm[30], *l2_tus_b = prm[31];

  // workspace carve
  const size_t U = (size_t)NU * 128;   // user-sized buffer (floats)
  const size_t I = (size_t)NI * 128;   // item-sized buffer (floats)
  float* wsf = (float*)d_ws;
  float* A  = wsf;
  float* B  = A + U;
  float* C  = B + U;
  float* Dm = C + U;
  float* Em = Dm + U;
  float* F  = Em + U;
  float* P  = F + U;            // item-sized
  float* Q  = P + I;            // item-sized
  _Float16* wswz = (_Float16*)(Q + I);   // 64 KB weight swizzle staging

  float* OUT = (float*)d_out;   // mean at OUT[0..U), sigma at OUT[U..2U)

  // ----- Layer 1: DGCNLayer_cross -----
  // source chain: VU then UV
  run_gemm(s_fea, nullptr, l1_gc1_W, nullptr, A, NU, 128, ACT_NONE, wswz, stream);
  run_scatter_gcn(se_i, se_u, se_v, Es, A, P, NI, l1_gc1_b, ACT_LEAKY, stream);
  run_gemm(P, nullptr, l1_gc3_W, nullptr, Q, NI, 128, ACT_NONE, wswz, stream);
  run_scatter_gcn(se_u, se_i, se_v, Es, Q, B, NU, l1_gc3_b, ACT_LEAKY, stream);   // s_ho
  // target chain
  run_gemm(t_fea, nullptr, l1_gc2_W, nullptr, A, NU, 128, ACT_NONE, wswz, stream);
  run_scatter_gcn(te_i, te_u, te_v, Et, A, P, NI, l1_gc2_b, ACT_LEAKY, stream);
  run_gemm(P, nullptr, l1_gc4_W, nullptr, Q, NI, 128, ACT_NONE, wswz, stream);
  run_scatter_gcn(te_u, te_i, te_v, Et, Q, C, NU, l1_gc4_b, ACT_LEAKY, stream);   // t_ho
  // s_User / t_User / fused
  run_gemm(B, s_fea, l1_su_W, l1_su_b, Dm, NU, 256, ACT_RELU, wswz, stream);
  run_gemm(C, t_fea, l1_tu_W, l1_tu_b, Em, NU, 256, ACT_RELU, wswz, stream);
  run_fuse(Dm, Em, F, (long long)U, stream);                                       // su == tu

  // ----- Layer 2: LastLayer_cross -----
  // source
  run_gemm(F, nullptr, l2_gc1_W, nullptr, A, NU, 128, ACT_NONE, wswz, stream);
  run_scatter_gcn(se_i, se_u, se_v, Es, A, P, NI, l2_gc1_b, ACT_LEAKY, stream);    // s_ho
  run_gemm(P, nullptr, l2_gc3m_W, nullptr, Q, NI, 128, ACT_NONE, wswz, stream);
  run_scatter_gcn(se_u, se_i, se_v, Es, Q, B, NU, l2_gc3m_b, ACT_LEAKY, stream);   // s_mean
  run_gemm(P, nullptr, l2_gc3s_W, nullptr, Q, NI, 128, ACT_NONE, wswz, stream);
  run_scatter_gcn(se_u, se_i, se_v, Es, Q, C, NU, l2_gc3s_b, ACT_LEAKY, stream);   // s_lstd
  // target
  run_gemm(F, nullptr, l2_gc2_W, nullptr, A, NU, 128, ACT_NONE, wswz, stream);
  run_scatter_gcn(te_i, te_u, te_v, Et, A, P, NI, l2_gc2_b, ACT_LEAKY, stream);    // t_ho
  run_gemm(P, nullptr, l2_gc4m_W, nullptr, Q, NI, 128, ACT_NONE, wswz, stream);
  run_scatter_gcn(te_u, te_i, te_v, Et, Q, Dm, NU, l2_gc4m_b, ACT_LEAKY, stream);  // t_mean
  run_gemm(P, nullptr, l2_gc4s_W, nullptr, Q, NI, 128, ACT_NONE, wswz, stream);
  run_scatter_gcn(te_u, te_i, te_v, Et, Q, Em, NU, l2_gc4s_b, ACT_LEAKY, stream);  // t_lstd

  // heads: mean
  run_gemm(B,  F, l2_sum_W, l2_sum_b, A, NU, 256, ACT_NONE, wswz, stream);  // s_m
  run_gemm(Dm, F, l2_tum_W, l2_tum_b, B, NU, 256, ACT_NONE, wswz, stream);  // t_m
  run_fuse(A, B, OUT, (long long)U, stream);
  // heads: sigma
  run_gemm(C,  F, l2_sus_W, l2_sus_b, A, NU, 256, ACT_NONE, wswz, stream);  // s_s
  run_gemm(Em, F, l2_tus_W, l2_tus_b, B, NU, 256, ACT_NONE, wswz, stream);  // t_s
  run_fuse(A, B, OUT + U, (long long)U, stream);
}
